// UBSN_1425929142281
// MI455X (gfx1250) — compile-verified, matrix operand order checked by reading, not run on hardware
//
#include <hip/hip_runtime.h>
#include <hip/hip_bf16.h>

typedef __attribute__((ext_vector_type(16))) _Float16 v16h;
typedef __attribute__((ext_vector_type(8)))  _Float16 v8h;
typedef __attribute__((ext_vector_type(8)))  float    v8f;

// ============================ pd_down (f32) ============================
__global__ __launch_bounds__(256) void k_pd_down(const float* __restrict__ x,
                                                 float* __restrict__ xpd) {
  int i = blockIdx.x * 256 + threadIdx.x;
  if (i >= 384 * 384) return;
  int yy = i / 384, xx = i % 384;
  xpd[i] = x[((yy % 192) * 2 + yy / 192) * 384 + ((xx % 192) * 2 + xx / 192)];
}

// ================= first conv 1->128 (direct, cheap) ==================
__global__ __launch_bounds__(256) void k_conv_nb0(const float* __restrict__ xpd,
                                                  const float* __restrict__ w,
                                                  const float* __restrict__ bias,
                                                  _Float16* __restrict__ out) {
  int id = blockIdx.x * 256 + threadIdx.x;
  if (id >= 384 * 384 * 16) return;
  int grp = id & 15;           // 8-channel group
  int p   = id >> 4;
  int y = p / 384, x = p % 384;
  float acc[8] = {0.f, 0.f, 0.f, 0.f, 0.f, 0.f, 0.f, 0.f};
#pragma unroll
  for (int t = 0; t < 9; ++t) {
    int ys = y + t / 3 - 1, xs = x + t % 3 - 1;
    if (ys < 0 || ys >= 384 || xs < 0 || xs >= 384) continue;
    float v = xpd[ys * 384 + xs];
#pragma unroll
    for (int c = 0; c < 8; ++c) acc[c] += w[(grp * 8 + c) * 9 + t] * v;
  }
  v8h o;
#pragma unroll
  for (int c = 0; c < 8; ++c)
    o[c] = (_Float16)fmaxf(acc[c] + bias[grp * 8 + c], 0.f);
  *(v8h*)(out + (size_t)p * 128 + grp * 8) = o;
}

// ===== pack OIHW f32 weights into WMMA A-fragment order (f16) =====
// dst[((t*kchunks+kc)*8+m)*512 + lane*16 + e]
//   lane<16: M=m*16+lane,   e<8 -> K=kc*32+e,    e>=8 -> K=kc*32+16+(e-8)
//   lane>=16: M=m*16+lane-16, K offsets +8 (per ISA 16-bit A layout)
__global__ __launch_bounds__(256) void k_pack_w(const float* __restrict__ src,
                                                _Float16* __restrict__ dst,
                                                int kchunks, int hole) {
  int id = blockIdx.x * 256 + threadIdx.x;
  int total = 9 * kchunks * 4096;
  if (id >= total) return;
  int e    = id & 15;
  int lane = (id >> 4) & 31;
  int m    = (id >> 9) & 7;
  int tk   = id >> 12;
  int kc   = tk % kchunks;
  int t    = tk / kchunks;
  int Cin  = kchunks * 32;
  int co   = m * 16 + (lane & 15);
  int off  = (lane >= 16) ? 8 : 0;
  int k    = (e < 8) ? (off + e) : (16 + off + (e - 8));
  int ci   = kc * 32 + k;
  float v  = src[((size_t)co * Cin + ci) * 9 + t];
  if (hole && t == 4) v = 0.f;     // blind-spot: zero center tap
  dst[id] = (_Float16)v;
}

// =================== implicit-GEMM 3x3 conv via WMMA ===================
// in:  NHWC f16 (upmode: half-res u, channels 0..127)
// in2: NHWC f16 (upmode: full-res skip, channels 128..255)
// out: NHWC f16, 128 channels, ReLU applied.
// Workgroup tile: 128(M out-ch) x 128(N pixels); 8 waves, each 32x64
// (2 M-frags x 4 N-frags = 8 WMMA accumulators).
// B tile double-buffered in LDS, staged by async-to-LDS DMA (ASYNCcnt);
// tap/channel cursor advances incrementally (no div/mod in the loop).
__global__ __launch_bounds__(256) void k_conv3_wmma(
    const _Float16* __restrict__ in, const _Float16* __restrict__ in2,
    const _Float16* __restrict__ wpk, const float* __restrict__ bias,
    _Float16* __restrict__ out, int H, int W, int kchunks, int dil, int upmode)
{
  __shared__ __align__(16) _Float16 lds[2][128 * 32];  // [buf][pixel][ci], 2x8KB
  const int npix  = H * W;
  const int tid   = threadIdx.x;
  const int lane  = tid & 31;
  const int wave  = tid >> 5;
  const int mblk  = (wave & 3) * 32;        // 4 waves across M
  const int nhalf = (wave >> 2) * 64;       // 2 waves across N
  const int pbase = blockIdx.x * 128;

  v8f acc0[4] = {};   // mfrag 0, nfrag 0..3
  v8f acc1[4] = {};   // mfrag 1, nfrag 0..3

  // staging role: thread covers pixel lp, 16 channels cg16.. (2x16B per step)
  const int lp   = tid >> 1;                // 0..127
  const int cg16 = (tid & 1) * 16;
  int ps = pbase + lp; if (ps >= npix) ps = npix - 1;   // clamp (store masked later)
  const int py = ps / W, px = ps % W;
  const int Wh = W >> 1;
  const int S  = 9 * kchunks;               // total K steps

  // incremental staging cursor: tap (dy,dx) starts at (-1,-1)*dil
  int st_kc = 0, st_tx = 0;
  int st_ys = py - dil, st_xs = px - dil;

  auto stage = [&](int buf) {
    bool inb = (st_ys >= 0) && (st_ys < H) && (st_xs >= 0) && (st_xs < W);
    _Float16* dst = &lds[buf][lp * 32 + cg16];
    if (inb) {
      int cg = st_kc * 32 + cg16;
      const _Float16* src;
      if (!upmode) {
        src = in + ((size_t)st_ys * W + st_xs) * 128 + cg;
      } else if (cg < 128) {      // nearest-neighbor up2 of half-res u
        src = in + ((size_t)(st_ys >> 1) * Wh + (st_xs >> 1)) * 128 + cg;
      } else {                    // skip connection (full res)
        src = in2 + ((size_t)st_ys * W + st_xs) * 128 + (cg - 128);
      }
      unsigned laddr = (unsigned)(unsigned long long)dst;     // LDS byte offset
      unsigned long long gaddr = (unsigned long long)src;
      // 2x16B per lane, memory -> LDS with no VGPR round trip (ASYNCcnt).
      // INST_OFFSET applies to both global and LDS address (ISA 15.18).
      asm volatile("global_load_async_to_lds_b128 %0, %1, off"
                   :: "v"(laddr), "v"(gaddr) : "memory");
      asm volatile("global_load_async_to_lds_b128 %0, %1, off offset:16"
                   :: "v"(laddr), "v"(gaddr) : "memory");
    } else {
      v8h z = {};
      *(v8h*)dst = z;             // zero padding for out-of-bounds taps
      *(v8h*)(dst + 8) = z;
    }
  };
  auto advance = [&]() {          // next (kc, tap) without div/mod
    if (++st_kc == kchunks) {
      st_kc = 0;
      st_xs += dil;
      if (++st_tx == 3) { st_tx = 0; st_xs -= 3 * dil; st_ys += dil; }
    }
  };

  stage(0);
  advance();
  asm volatile("s_wait_asynccnt 0x0" ::: "memory");
  __syncthreads();

  // A-fragment cursor: advances by one 32-K slice (4096 halves) per step
  const _Float16* aptr = wpk + (size_t)(mblk >> 4) * 512 + lane * 16;

  for (int s = 0; s < S; ++s) {
    const int buf = s & 1;
    if (s + 1 < S) { stage(buf ^ 1); advance(); }   // async fill of next tile

    const v16h a0 = *(const v16h*)(aptr);
    const v16h a1 = *(const v16h*)(aptr + 512);
    aptr += 4096;

    // B fragments: two contiguous 16B LDS runs per lane (ISA 16-bit layout)
    const int hof = (lane >= 16) ? 8 : 0;
    const _Float16* lb = &lds[buf][0];
    const int rbase = (nhalf + (lane & 15)) * 32;
#pragma unroll
    for (int nf = 0; nf < 4; ++nf) {
      const int r = rbase + nf * 16 * 32;
      v8h blo = *(const v8h*)&lb[r + hof];
      v8h bhi = *(const v8h*)&lb[r + 16 + hof];
      v16h b = __builtin_shufflevector(blo, bhi, 0,1,2,3,4,5,6,7,8,9,10,11,12,13,14,15);
      acc0[nf] = __builtin_amdgcn_wmma_f32_16x16x32_f16(false, a0, false, b, (short)0, acc0[nf], false, false);
      acc1[nf] = __builtin_amdgcn_wmma_f32_16x16x32_f16(false, a1, false, b, (short)0, acc1[nf], false, false);
    }

    asm volatile("s_wait_asynccnt 0x0" ::: "memory");  // next tile landed
    __syncthreads();                                   // one barrier per step
  }

  // epilogue: bias + ReLU + f16, lane holds 8 consecutive channels per frag
  const int hof  = (lane >= 16) ? 8 : 0;
  const int nloc = lane & 15;
  const int cb0  = mblk + hof;
  const int cb1  = mblk + 16 + hof;
  float bi0[8], bi1[8];
#pragma unroll
  for (int v = 0; v < 8; ++v) { bi0[v] = bias[cb0 + v]; bi1[v] = bias[cb1 + v]; }
#pragma unroll
  for (int nf = 0; nf < 4; ++nf) {
    const int n = pbase + nhalf + nf * 16 + nloc;
    if (n < npix) {
      v8h o0, o1;
#pragma unroll
      for (int v = 0; v < 8; ++v) {
        o0[v] = (_Float16)fmaxf(acc0[nf][v] + bi0[v], 0.f);
        o1[v] = (_Float16)fmaxf(acc1[nf][v] + bi1[v], 0.f);
      }
      *(v8h*)(out + (size_t)n * 128 + cb0) = o0;
      *(v8h*)(out + (size_t)n * 128 + cb1) = o1;
    }
  }
}

// ============================ maxpool 2x2 =============================
__global__ __launch_bounds__(256) void k_maxpool2(const _Float16* __restrict__ in,
                                                  _Float16* __restrict__ out,
                                                  int Ho, int Wo) {
  int id = blockIdx.x * 256 + threadIdx.x;
  int total = Ho * Wo * 16;
  if (id >= total) return;
  int grp = id & 15;
  int op  = id >> 4;
  int y = op / Wo, x = op % Wo;
  int Wi = Wo * 2;
  const _Float16* b = in + ((size_t)(2 * y) * Wi + 2 * x) * 128 + grp * 8;
  v8h a0 = *(const v8h*)b;
  v8h a1 = *(const v8h*)(b + 128);
  v8h a2 = *(const v8h*)(b + (size_t)Wi * 128);
  v8h a3 = *(const v8h*)(b + (size_t)Wi * 128 + 128);
  v8h m;
#pragma unroll
  for (int i = 0; i < 8; ++i) {
    _Float16 u = a0[i] > a1[i] ? a0[i] : a1[i];
    _Float16 v = a2[i] > a3[i] ? a2[i] : a3[i];
    m[i] = u > v ? u : v;
  }
  *(v8h*)(out + (size_t)op * 128 + grp * 8) = m;
}

// ========== fused 1x1 head 128->48->24->8->1 + pd_up (f32 out) =========
__global__ __launch_bounds__(256) void k_out_chain(
    const _Float16* __restrict__ u,
    const float* __restrict__ w0, const float* __restrict__ b0,
    const float* __restrict__ w1, const float* __restrict__ b1,
    const float* __restrict__ w2, const float* __restrict__ b2,
    const float* __restrict__ w3, const float* __restrict__ b3,
    float* __restrict__ out)
{
  __shared__ __align__(16) _Float16 sx[256 * 128];   // per-thread pixel features
  int tid = threadIdx.x;
  int p = blockIdx.x * 256 + tid;
  int pc = p < 384 * 384 ? p : 384 * 384 - 1;
  const _Float16* up = u + (size_t)pc * 128;
#pragma unroll
  for (int g = 0; g < 16; ++g)
    *(v8h*)&sx[tid * 128 + g * 8] = *(const v8h*)(up + g * 8);
  const _Float16* xin = &sx[tid * 128];

  float y0[48];
  for (int o = 0; o < 48; ++o) {
    float s = b0[o];
    for (int c = 0; c < 128; ++c) s += w0[o * 128 + c] * (float)xin[c];
    y0[o] = fmaxf(s, 0.f);
  }
  float y1[24];
  for (int o = 0; o < 24; ++o) {
    float s = b1[o];
    for (int c = 0; c < 48; ++c) s += w1[o * 48 + c] * y0[c];
    y1[o] = fmaxf(s, 0.f);
  }
  float y2[8];
  for (int o = 0; o < 8; ++o) {
    float s = b2[o];
    for (int c = 0; c < 24; ++c) s += w2[o * 24 + c] * y1[c];
    y2[o] = fmaxf(s, 0.f);
  }
  float y3 = b3[0];
  for (int c = 0; c < 8; ++c) y3 += w3[c] * y2[c];

  if (p < 384 * 384) {
    int yy = p / 384, xx = p % 384;
    out[((yy % 192) * 2 + yy / 192) * 384 + ((xx % 192) * 2 + xx / 192)] = y3;
  }
}

// ============================= launcher ===============================
extern "C" void kernel_launch(void* const* d_in, const int* in_sizes, int n_in,
                              void* d_out, int out_size, void* d_ws, size_t ws_size,
                              hipStream_t stream)
{
  const float* x     = (const float*)d_in[0];
  const float* nbw0  = (const float*)d_in[1];
  const float* nbb0  = (const float*)d_in[2];
  const float* nbw1  = (const float*)d_in[3];
  const float* nbb1  = (const float*)d_in[4];
  const float* encw  = (const float*)d_in[5];
  const float* encb  = (const float*)d_in[6];
  const float* deepw = (const float*)d_in[7];
  const float* deepb = (const float*)d_in[8];
  const float* blw   = (const float*)d_in[9];
  const float* blb   = (const float*)d_in[10];
  const float* upw   = (const float*)d_in[11];
  const float* upb   = (const float*)d_in[12];
  const float* ow0 = (const float*)d_in[13]; const float* ob0 = (const float*)d_in[14];
  const float* ow1 = (const float*)d_in[15]; const float* ob1 = (const float*)d_in[16];
  const float* ow2 = (const float*)d_in[17]; const float* ob2 = (const float*)d_in[18];
  const float* ow3 = (const float*)d_in[19]; const float* ob3 = (const float*)d_in[20];
  float* out = (float*)d_out;

  char* ws = (char*)d_ws;
  size_t off = 0;
  auto alloc = [&](size_t bytes) -> char* {
    char* p = ws + off;
    off = (off + bytes + 255) & ~(size_t)255;
    return p;
  };
  auto actsz = [](int h, int w) { return (size_t)h * w * 128 * 2; };

  float*     xpd = (float*)alloc(384 * 384 * 4);
  _Float16*  T0  = (_Float16*)alloc(actsz(384, 384));
  _Float16*  T1  = (_Float16*)alloc(actsz(384, 384));
  _Float16*  BL0 = (_Float16*)alloc(actsz(384, 384));
  _Float16*  M1  = (_Float16*)alloc(actsz(192, 192));
  _Float16*  G1  = (_Float16*)alloc(actsz(192, 192));
  _Float16*  M2  = (_Float16*)alloc(actsz(96, 96));
  _Float16*  G2  = (_Float16*)alloc(actsz(96, 96));
  _Float16*  M3  = (_Float16*)alloc(actsz(48, 48));
  _Float16*  G3  = (_Float16*)alloc(actsz(48, 48));
  _Float16*  M4  = (_Float16*)alloc(actsz(24, 24));
  _Float16*  G4  = (_Float16*)alloc(actsz(24, 24));
  _Float16*  M5  = (_Float16*)alloc(actsz(12, 12));
  _Float16*  G5  = (_Float16*)alloc(actsz(12, 12));

  _Float16* wnb1 = (_Float16*)alloc(147456 * 2);
  _Float16* wenc[5];  for (int i = 0; i < 5; ++i) wenc[i] = (_Float16*)alloc(147456 * 2);
  _Float16* wdeep[2]; for (int i = 0; i < 2; ++i) wdeep[i] = (_Float16*)alloc(147456 * 2);
  _Float16* wbl[6];   for (int i = 0; i < 6; ++i) wbl[i]  = (_Float16*)alloc(147456 * 2);
  _Float16* wup[5];   for (int i = 0; i < 5; ++i) wup[i]  = (_Float16*)alloc(294912 * 2);

  auto pack = [&](const float* src, _Float16* dst, int kch, int hole) {
    int total = 9 * kch * 4096;
    k_pack_w<<<(total + 255) / 256, 256, 0, stream>>>(src, dst, kch, hole);
  };
  auto conv = [&](const _Float16* in, const _Float16* in2, const _Float16* w,
                  const float* bias, _Float16* o, int H, int W, int kch, int dil, int up) {
    int npix = H * W;
    k_conv3_wmma<<<(npix + 127) / 128, 256, 0, stream>>>(in, in2, w, bias, o, H, W, kch, dil, up);
  };
  auto pool = [&](const _Float16* in, _Float16* o, int Ho, int Wo) {
    int total = Ho * Wo * 16;
    k_maxpool2<<<(total + 255) / 256, 256, 0, stream>>>(in, o, Ho, Wo);
  };

  // weight packing (all f16, hole mask folded into blind weights)
  pack(nbw1, wnb1, 4, 0);
  for (int d = 0; d < 5; ++d) pack(encw + (size_t)d * 128 * 128 * 9, wenc[d], 4, 0);
  for (int i = 0; i < 2; ++i) pack(deepw + (size_t)i * 128 * 128 * 9, wdeep[i], 4, 0);
  for (int d = 0; d < 6; ++d) pack(blw + (size_t)d * 128 * 128 * 9, wbl[d], 4, 1);
  for (int i = 0; i < 5; ++i) pack(upw + (size_t)i * 128 * 256 * 9, wup[i], 8, 0);

  // stem
  k_pd_down<<<(384 * 384 + 255) / 256, 256, 0, stream>>>(x, xpd);
  k_conv_nb0<<<(384 * 384 * 16 + 255) / 256, 256, 0, stream>>>(xpd, nbw0, nbb0, T0);

  // encoder
  conv(T0, nullptr, wnb1,    nbb1,       T1, 384, 384, 4, 1, 0);  // F0a
  pool(T1, M1, 192, 192);
  conv(M1, nullptr, wenc[0], encb + 0,   G1, 192, 192, 4, 1, 0);  // F1a
  pool(G1, M2, 96, 96);
  conv(M2, nullptr, wenc[1], encb + 128, G2, 96, 96, 4, 1, 0);    // F2
  pool(G2, M3, 48, 48);
  conv(M3, nullptr, wenc[2], encb + 256, G3, 48, 48, 4, 1, 0);    // F3
  pool(G3, M4, 24, 24);
  conv(M4, nullptr, wenc[3], encb + 384, G4, 24, 24, 4, 1, 0);    // F4
  pool(G4, M5, 12, 12);
  conv(M5, nullptr, wenc[4], encb + 512, G5, 12, 12, 4, 1, 0);    // F5

  // deep convs (replace feats[0], feats[1])
  conv(T1, nullptr, wdeep[0], deepb + 0,   T0, 384, 384, 4, 1, 0); // F0
  conv(G1, nullptr, wdeep[1], deepb + 128, M1, 192, 192, 4, 1, 0); // F1

  // blind (hole) convs, dilations [4,4,3,3,3,3]
  conv(T0, nullptr, wbl[0], blb + 0,   BL0, 384, 384, 4, 4, 0);
  conv(M1, nullptr, wbl[1], blb + 128, G1,  192, 192, 4, 4, 0);
  conv(G2, nullptr, wbl[2], blb + 256, M2,  96, 96, 4, 3, 0);
  conv(G3, nullptr, wbl[3], blb + 384, M3,  48, 48, 4, 3, 0);
  conv(G4, nullptr, wbl[4], blb + 512, M4,  24, 24, 4, 3, 0);
  conv(G5, nullptr, wbl[5], blb + 640, M5,  12, 12, 4, 3, 0);

  // decoder: up2(u) ++ blind[d] -> conv 256->128 (concat fused via in/in2)
  conv(M5, M4,  wup[0], upb + 0,   G4, 24, 24, 8, 1, 1);   // u4
  conv(G4, M3,  wup[1], upb + 128, G3, 48, 48, 8, 1, 1);   // u3
  conv(G3, M2,  wup[2], upb + 256, G2, 96, 96, 8, 1, 1);   // u2
  conv(G2, G1,  wup[3], upb + 384, M1, 192, 192, 8, 1, 1); // u1
  conv(M1, BL0, wup[4], upb + 512, T1, 384, 384, 8, 1, 1); // u0

  // 1x1 head + pd_up
  k_out_chain<<<(384 * 384 + 255) / 256, 256, 0, stream>>>(
      T1, ow0, ob0, ow1, ob1, ow2, ob2, ow3, ob3, out);

  (void)in_sizes; (void)n_in; (void)out_size; (void)ws_size;
}
